// Tmp2_3modals_489626271913
// MI455X (gfx1250) — matrix operand order repaired and checked by reading, not compiled, not run on hardware
//
#include <hip/hip_runtime.h>
#include <hip/hip_bf16.h>

#define NTOK 16384
#define CDIM 384
#define PC   1152   // k(384) | q(384) | v(384)
#define HS   48
#define NH   8

typedef __attribute__((ext_vector_type(16))) __bf16 v16bf;
typedef __attribute__((ext_vector_type(8)))  float  v8f;

// Types for the async global->LDS builtin: param 0 is
// '__attribute__((vector_size(16))) int __device__ *' (AS1), per hipcc diagnostic.
typedef int v4i_vs __attribute__((vector_size(16)));
typedef __attribute__((address_space(1))) void*    gvoid_p;
typedef __attribute__((address_space(3))) void*    lvoid_p;
typedef __attribute__((address_space(1))) v4i_vs*  gv4i_p;
typedef __attribute__((address_space(3))) v4i_vs*  lv4i_p;

#if __has_builtin(__builtin_amdgcn_global_load_async_to_lds_b128)
#define HAVE_ASYNC_LDS 1
#else
#define HAVE_ASYNC_LDS 0
#endif

__device__ __forceinline__ void wait_async0() {
#if __has_builtin(__builtin_amdgcn_s_wait_asynccnt)
  __builtin_amdgcn_s_wait_asynccnt(0);
#else
  asm volatile("s_wait_asynccnt 0x0" ::: "memory");
#endif
}

__device__ __forceinline__ unsigned short f2bf(float x) {
  __bf16 h = (__bf16)x;                       // native v_cvt -> bf16 (RNE)
  return __builtin_bit_cast(unsigned short, h);
}
__device__ __forceinline__ float bf2f(unsigned short b) {
  return (float)__builtin_bit_cast(__bf16, b);
}

union FragU { v16bf v; unsigned u[8]; };

// A operand (16M x 32K). LDS tile stored row-major [m][k], k contiguous, pitch even (ushorts).
// Layout (ISA 7.12.2, 16-bit A 16x32): lane<16 -> K {0..7,16..23}, lane>=16 -> K {8..15,24..31}.
__device__ __forceinline__ v16bf load_frag_a(const unsigned short* base, int row0, int pitch, int k0) {
  int lane = threadIdx.x & 31;
  int hi = lane >> 4;
  const unsigned short* r = base + (size_t)(row0 + (lane & 15)) * pitch + k0;
  FragU f;
#pragma unroll
  for (int p = 0; p < 8; ++p) {
    int k = (p < 4) ? (hi * 8 + 2 * p) : (16 + hi * 8 + 2 * (p - 4));
    f.u[p] = *(const unsigned*)(r + k);
  }
  return f.v;
}
// B operand (32K x 16N). LDS tile stored row-major [n][k] (transposed), k contiguous.
// lane<16 -> K {0..7,16..23}, lane>=16 -> K {8..15,24..31} (same as A).
__device__ __forceinline__ v16bf load_frag_b(const unsigned short* base, int row0, int pitch, int k0) {
  int lane = threadIdx.x & 31;
  int hi = lane >> 4;
  const unsigned short* r = base + (size_t)(row0 + (lane & 15)) * pitch + k0 + hi * 8;
  FragU f;
#pragma unroll
  for (int j = 0; j < 8; ++j) f.u[j] = *(const unsigned*)(r + ((j < 4) ? 2 * j : 16 + 2 * (j - 4)));
  return f.v;
}

__device__ __forceinline__ v8f wmma_bf16(v16bf a, v16bf b, v8f c) {
  return __builtin_amdgcn_wmma_f32_16x16x32_bf16(false, a, false, b, (short)0, c, false, false);
}

// ------- K0: pack weights TRANSPOSED into bf16 Wt[var][j][c]  (j = sel*384 + h*48 + d) --------
__global__ __launch_bounds__(256) void k_pack_w(const float* Wk0, const float* Wk1,
                                                const float* Wq0, const float* Wq1,
                                                const float* Wv0, const float* Wv1,
                                                unsigned short* Wt) {
  int i = blockIdx.x * 256 + threadIdx.x;
  if (i >= 2 * CDIM * PC) return;
  int c = i % CDIM, j = (i / CDIM) % PC, var = i / (CDIM * PC);
  int sel = j / CDIM, hd = j % CDIM, h = hd / HS, d = hd % HS;
  const float* W;
  if (sel == 0)      W = var ? Wk1 : Wk0;
  else if (sel == 1) W = var ? Wq1 : Wq0;
  else               W = var ? Wv1 : Wv0;
  Wt[i] = f2bf(W[((size_t)h * CDIM + c) * HS + d]);   // W[h][c][d]
}

// ---------------- K1: P[bm][n][j] = sum_c x[b][c][n] * Wcat[c][j]  (bf16 out) -----------------
__global__ __launch_bounds__(256) void k_proj_gemm(const float* x0, const float* x1, const float* x2,
                                                   const unsigned short* Wt, unsigned short* P) {
  __shared__ __align__(16) unsigned short As[128][40];  // [token][k], pitch 40
  __shared__ __align__(16) unsigned short Bs[64][56];   // [j][k],     pitch 56 (16B-aligned chunks)
  int bm = blockIdx.z, mod = bm >> 1, b = bm & 1;
  const float* x = (mod == 0) ? x0 : ((mod == 1) ? x1 : x2);
  const float* xb = x + (size_t)b * CDIM * NTOK;
  const unsigned short* W = Wt + (size_t)(mod ? 1 : 0) * CDIM * PC;  // [j][c]
  unsigned short* Pp = P + (size_t)bm * NTOK * PC;
  int n0 = blockIdx.x * 128, j0 = blockIdx.y * 64;
  int tid = threadIdx.x, lane = tid & 31, wv = tid >> 5;
  int wm = wv >> 1, wn = wv & 1;
  v8f acc[2][2] = {};
  for (int kk = 0; kk < CDIM; kk += 32) {
    // --- B tile: straight 16B/lane copy, async global->LDS (ASYNCcnt path) ---
    {
      int j = tid >> 2, part = tid & 3;
      const unsigned short* g = W + (size_t)(j0 + j) * CDIM + kk + part * 8;
      unsigned short* l = &Bs[j][part * 8];
#if HAVE_ASYNC_LDS
      __builtin_amdgcn_global_load_async_to_lds_b128(
          (gv4i_p)(gvoid_p)(void*)const_cast<unsigned short*>(g),
          (lv4i_p)(lvoid_p)(void*)l, 0, 0);
#else
#pragma unroll
      for (int q = 0; q < 8; ++q) l[q] = g[q];
#endif
    }
    // --- A tile: f32 -> bf16 with transpose to [token][k] ---
    for (int i = tid; i < 128 * 32; i += 256) {
      int cc = i >> 7, tok = i & 127;
      As[tok][cc] = f2bf(xb[(size_t)(kk + cc) * NTOK + n0 + tok]);
    }
    if (kk + 32 < CDIM && tid < 32)                    // hint next A tile into cache
      __builtin_prefetch(xb + (size_t)(kk + 32 + tid) * NTOK + n0, 0, 1);
    wait_async0();
    __syncthreads();
    v16bf af0 = load_frag_a(&As[0][0], wm * 32, 40, 0);
    v16bf af1 = load_frag_a(&As[0][0], wm * 32 + 16, 40, 0);
    v16bf bf0 = load_frag_b(&Bs[0][0], wn * 32, 56, 0);
    v16bf bf1 = load_frag_b(&Bs[0][0], wn * 32 + 16, 56, 0);
    acc[0][0] = wmma_bf16(af0, bf0, acc[0][0]);
    acc[0][1] = wmma_bf16(af0, bf1, acc[0][1]);
    acc[1][0] = wmma_bf16(af1, bf0, acc[1][0]);
    acc[1][1] = wmma_bf16(af1, bf1, acc[1][1]);
    __syncthreads();
  }
  int hi = lane >> 4, ln = lane & 15;
#pragma unroll
  for (int mi = 0; mi < 2; ++mi)
#pragma unroll
    for (int ni = 0; ni < 2; ++ni)
#pragma unroll
      for (int r = 0; r < 8; ++r) {
        int row = n0 + wm * 32 + mi * 16 + r + 8 * hi;
        int col = j0 + wn * 32 + ni * 16 + ln;
        Pp[(size_t)row * PC + col] = f2bf(acc[mi][ni][r]);
      }
}

// ---------------- K2: per-column (token axis) softmax stats for k ------------------------------
__global__ __launch_bounds__(256) void k_colstats(const unsigned short* P, float2* stats) {
  int col = blockIdx.x;                       // bm*384 + j
  int bm = col / CDIM, j = col % CDIM;
  const unsigned short* Pp = P + (size_t)bm * NTOK * PC + j;
  __shared__ float red[256];
  int tid = threadIdx.x;
  float mx = -1e30f;
  for (int n = tid; n < NTOK; n += 256) mx = fmaxf(mx, bf2f(Pp[(size_t)n * PC]));
  red[tid] = mx; __syncthreads();
  for (int s = 128; s > 0; s >>= 1) { if (tid < s) red[tid] = fmaxf(red[tid], red[tid + s]); __syncthreads(); }
  mx = red[0]; __syncthreads();
  float sm = 0.0f;
  for (int n = tid; n < NTOK; n += 256) sm += __expf(bf2f(Pp[(size_t)n * PC]) - mx);
  red[tid] = sm; __syncthreads();
  for (int s = 128; s > 0; s >>= 1) { if (tid < s) red[tid] += red[tid + s]; __syncthreads(); }
  if (tid == 0) stats[col] = make_float2(mx, 1.0f / red[0]);
}

// ---------------- K3: w[d][e] = sum_n softmax_k[n,d] * v[n,e]  (split-K across 8 waves) -------
__global__ __launch_bounds__(256) void k_kv(const unsigned short* P, const float2* stats, float* Wpart) {
  __shared__ unsigned short Ak[8][48][40];   // per-wave [d][n]
  __shared__ unsigned short Bv[8][48][40];   // per-wave [e][n]
  int h = blockIdx.x, bm = blockIdx.y;
  int tid = threadIdx.x, lane = tid & 31, wv = tid >> 5;
  const unsigned short* Pp = P + (size_t)bm * NTOK * PC;
  const float2* st = stats + bm * CDIM + h * HS;
  v8f acc[3][3] = {};
  for (int blk = 0; blk < 64; ++blk) {
    int nb = wv * 2048 + blk * 32;
    for (int i = lane; i < 48 * 32; i += 32) {
      int d = i >> 5, n = i & 31;
      size_t row = (size_t)(nb + n) * PC;
      float kl = bf2f(Pp[row + h * HS + d]);
      float2 s = st[d];
      Ak[wv][d][n] = f2bf(__expf(kl - s.x) * s.y);
      Bv[wv][d][n] = Pp[row + 2 * CDIM + h * HS + d];
    }
    asm volatile("" ::: "memory");           // wave-private LDS; in-order DS pipe
#pragma unroll
    for (int mt = 0; mt < 3; ++mt) {
      v16bf a = load_frag_a(&Ak[wv][0][0], mt * 16, 40, 0);
#pragma unroll
      for (int nt = 0; nt < 3; ++nt) {
        v16bf bb = load_frag_b(&Bv[wv][0][0], nt * 16, 40, 0);
        acc[mt][nt] = wmma_bf16(a, bb, acc[mt][nt]);
      }
    }
    asm volatile("" ::: "memory");
  }
  float* Wp = Wpart + ((size_t)(bm * NH + h) * 8 + wv) * (HS * HS);
  int hi = lane >> 4, ln = lane & 15;
#pragma unroll
  for (int mt = 0; mt < 3; ++mt)
#pragma unroll
    for (int nt = 0; nt < 3; ++nt)
#pragma unroll
      for (int r = 0; r < 8; ++r)
        Wp[(mt * 16 + r + 8 * hi) * HS + nt * 16 + ln] = acc[mt][nt][r];
}

__global__ __launch_bounds__(256) void k_kv_reduce(const float* Wpart, float* Wmat) {
  int i = blockIdx.x * 256 + threadIdx.x;
  if (i >= 48 * HS * HS) return;
  int grp = i / (HS * HS), idx = i % (HS * HS);
  const float* p = Wpart + (size_t)grp * 8 * (HS * HS);
  float s = 0.0f;
#pragma unroll
  for (int w = 0; w < 8; ++w) s += p[w * (HS * HS) + idx];
  Wmat[i] = s;
}

// ---------------- K4: a = softmax_row(q) @ (w_o1 + w_o2); merge heads + residual --------------
__global__ __launch_bounds__(256) void k_qw(const unsigned short* P, const float* Wmat,
                                            const float* x0, const float* x1, const float* x2,
                                            float* out) {
  __shared__ unsigned short Wsum[48][66];    // [e][d], d padded to 64 with zeros
  __shared__ unsigned short Qh[8][16][66];   // per-wave [token][d], padded
  int bm = blockIdx.y, mod = bm >> 1, b = bm & 1;
  int tid = threadIdx.x, lane = tid & 31, wv = tid >> 5;
  int o1 = (mod + 1) % 3, o2 = (mod + 2) % 3;
  const float* W1 = Wmat + (size_t)((o1 * 2 + b) * NH) * (HS * HS);
  const float* W2 = Wmat + (size_t)((o2 * 2 + b) * NH) * (HS * HS);
  const unsigned short* Pp = P + (size_t)bm * NTOK * PC;
  const float* x = (mod == 0) ? x0 : ((mod == 1) ? x1 : x2);
  const float* xb = x + (size_t)b * CDIM * NTOK;
  float* ob = out + (size_t)bm * CDIM * NTOK;
  int tb = blockIdx.x * 128 + wv * 16;

  for (int h = 0; h < NH; ++h) {
    for (int i = tid; i < 48 * 64; i += 256) {         // stage transposed w-sum
      int d = i & 63, e = i >> 6;
      unsigned short v = 0;
      if (d < 48) v = f2bf(W1[h * (HS * HS) + d * HS + e] + W2[h * (HS * HS) + d * HS + e]);
      Wsum[e][d] = v;
    }
    if (lane < 16) {                                   // row softmax of q (48 wide)
      const unsigned short* qr = Pp + (size_t)(tb + lane) * PC + CDIM + h * HS;
      float vals[HS];
      float mx = -1e30f;
#pragma unroll
      for (int d = 0; d < HS; ++d) { vals[d] = bf2f(qr[d]); mx = fmaxf(mx, vals[d]); }
      float s = 0.0f;
#pragma unroll
      for (int d = 0; d < HS; ++d) { vals[d] = __expf(vals[d] - mx); s += vals[d]; }
      float rs = 1.0f / s;
#pragma unroll
      for (int d = 0; d < HS; ++d) Qh[wv][lane][d] = f2bf(vals[d] * rs);
#pragma unroll
      for (int d = HS; d < 64; ++d) Qh[wv][lane][d] = 0;
    }
    __syncthreads();
    v8f acc[3] = {};
#pragma unroll
    for (int ks = 0; ks < 2; ++ks) {                   // K=48 padded to 64: two k-steps
      v16bf a = load_frag_a(&Qh[wv][0][0], 0, 66, ks * 32);
#pragma unroll
      for (int nt = 0; nt < 3; ++nt) {
        v16bf bb = load_frag_b(&Wsum[0][0], nt * 16, 66, ks * 32);
        acc[nt] = wmma_bf16(a, bb, acc[nt]);
      }
    }
    int hi = lane >> 4, ln = lane & 15;
#pragma unroll
    for (int nt = 0; nt < 3; ++nt)
#pragma unroll
      for (int r = 0; r < 8; ++r) {
        int e = nt * 16 + ln;
        int tok = tb + r + 8 * hi;
        size_t idx = (size_t)(h * HS + e) * NTOK + tok;
        ob[idx] = acc[nt][r] + xb[idx];                // residual, [b][c][n] layout
      }
    __syncthreads();
  }
}

// ---------------- K5: InstanceNorm3d (affine=False, biased var) in place ----------------------
__global__ __launch_bounds__(256) void k_inorm(float* out) {
  size_t base = (size_t)blockIdx.x * NTOK;
  __shared__ float r1[256], r2[256];
  int tid = threadIdx.x;
  float s = 0.0f, s2 = 0.0f;
  for (int n = tid; n < NTOK; n += 256) { float v = out[base + n]; s += v; s2 += v * v; }
  r1[tid] = s; r2[tid] = s2; __syncthreads();
  for (int st = 128; st > 0; st >>= 1) {
    if (tid < st) { r1[tid] += r1[tid + st]; r2[tid] += r2[tid + st]; }
    __syncthreads();
  }
  float mean = r1[0] * (1.0f / NTOK);
  float var  = r2[0] * (1.0f / NTOK) - mean * mean;
  float sc = rsqrtf(var + 1e-5f);
  for (int n = tid; n < NTOK; n += 256) out[base + n] = (out[base + n] - mean) * sc;
}

extern "C" void kernel_launch(void* const* d_in, const int* in_sizes, int n_in,
                              void* d_out, int out_size, void* d_ws, size_t ws_size,
                              hipStream_t stream) {
  const float* x0  = (const float*)d_in[0];
  const float* x1  = (const float*)d_in[1];
  const float* x2  = (const float*)d_in[2];
  const float* Wk0 = (const float*)d_in[3];
  const float* Wk1 = (const float*)d_in[4];
  const float* Wq0 = (const float*)d_in[5];
  const float* Wq1 = (const float*)d_in[6];
  const float* Wv0 = (const float*)d_in[7];
  const float* Wv1 = (const float*)d_in[8];
  float* out = (float*)d_out;
  char* ws = (char*)d_ws;

  size_t oWb = 0;                                             // 2*1152*384*2  = 1.77 MB
  size_t oP  = 2ull * 1024 * 1024;                            // bf16 logits
  size_t szP = (size_t)6 * NTOK * PC * sizeof(unsigned short);// 226.5 MB
  size_t oSt = oP + szP;
  size_t szSt = (size_t)6 * CDIM * sizeof(float2);            // 18 KB
  size_t oWp = (oSt + szSt + 255) & ~(size_t)255;
  size_t szWp = (size_t)48 * 8 * HS * HS * sizeof(float);     // 3.5 MB
  size_t oWm = (oWp + szWp + 255) & ~(size_t)255;
  size_t total = oWm + (size_t)48 * HS * HS * sizeof(float);
  if (ws_size < total) return;                                // avoid OOB if scratch too small

  unsigned short* Wt   = (unsigned short*)(ws + oWb);
  unsigned short* P    = (unsigned short*)(ws + oP);
  float2*         stat = (float2*)(ws + oSt);
  float*          Wpart= (float*)(ws + oWp);
  float*          Wmat = (float*)(ws + oWm);

  k_pack_w   <<<(2 * CDIM * PC + 255) / 256, 256, 0, stream>>>(Wk0, Wk1, Wq0, Wq1, Wv0, Wv1, Wt);
  k_proj_gemm<<<dim3(NTOK / 128, PC / 64, 6), 256, 0, stream>>>(x0, x1, x2, Wt, P);
  k_colstats <<<6 * CDIM, 256, 0, stream>>>(P, stat);
  k_kv       <<<dim3(NH, 6), 256, 0, stream>>>(P, stat, Wpart);
  k_kv_reduce<<<(48 * HS * HS + 255) / 256, 256, 0, stream>>>(Wpart, Wmat);
  k_qw       <<<dim3(NTOK / 128, 6), 256, 0, stream>>>(P, Wmat, x0, x1, x2, out);
  k_inorm    <<<6 * CDIM, 256, 0, stream>>>(out);
}
